// LoFTREncoderLayer_21500606284118
// MI455X (gfx1250) — matrix operand order repaired and checked by reading, not compiled, not run on hardware
//
#include <hip/hip_runtime.h>
#include <cstddef>
#include <cstdint>

// ---------------------------------------------------------------------------
// Types / helpers
// ---------------------------------------------------------------------------
typedef __bf16 bf16_t;
typedef __attribute__((ext_vector_type(16))) __bf16 v16bf;
typedef __attribute__((ext_vector_type(8)))  __bf16 v8bf;
typedef __attribute__((ext_vector_type(8)))  float  v8f;

#define DEV static __device__ __forceinline__

DEV bf16_t f2bf(float f) {                       // round-to-nearest-even f32 -> bf16
  unsigned u = __builtin_bit_cast(unsigned, f);
  unsigned r = u + 0x7FFFu + ((u >> 16) & 1u);
  unsigned short h = (unsigned short)(r >> 16);
  return __builtin_bit_cast(bf16_t, h);
}
DEV float bf2f(bf16_t b) {
  unsigned short h = __builtin_bit_cast(unsigned short, b);
  unsigned u = ((unsigned)h) << 16;
  return __builtin_bit_cast(float, u);
}
DEV v8f vzero8() {
  v8f z;
#pragma unroll
  for (int i = 0; i < 8; ++i) z[i] = 0.f;
  return z;
}

// A fragment (16x32, M x K) from row-major memory A[m,k], leading dim ld.
// Order used here (same as B): lane l -> m = l&15 ; 16 halves = K (l>>4)*16 + 0..15 (contiguous).
DEV v16bf loadA_rm(const bf16_t* __restrict__ A, int ld, int row0, int k0) {
  int l = threadIdx.x & 31;
  const bf16_t* p = A + (size_t)(row0 + (l & 15)) * ld + k0 + ((l >> 4) << 4);
  return *(const v16bf*)p;
}

// B fragment (32x16, K x N) where B = W^T, W row-major [n, k] (ld elems).
// Order used here (same as A): lane l -> n = l&15 ; 16 halves = K (l>>4)*16 + 0..15 (contiguous).
DEV v16bf loadB_rm(const bf16_t* __restrict__ W, int ld, int n0, int k0) {
  int l = threadIdx.x & 31;
  const bf16_t* p = W + (size_t)(n0 + (l & 15)) * ld + k0 + ((l >> 4) << 4);
  return *(const v16bf*)p;
}

// A fragment from k-major storage: element (m,k) at base[k*stride + col0 + m].
DEV v16bf loadA_km(const bf16_t* base, int stride, int col0) {
  int l = threadIdx.x & 31;
  int m  = col0 + (l & 15);
  int kb = (l >> 4) << 4;
  v16bf f;
#pragma unroll
  for (int j = 0; j < 8; ++j) {
    f[j]     = base[(kb + j) * stride + m];
    f[j + 8] = base[(kb + 8 + j) * stride + m];
  }
  return f;
}

// B fragment from k-major storage: element (k,n) at base[k*stride + col0 + n].
DEV v16bf loadB_km(const bf16_t* base, int stride, int col0) {
  int l = threadIdx.x & 31;
  int n  = col0 + (l & 15);
  int kb = (l >> 4) << 4;
  v16bf f;
#pragma unroll
  for (int j = 0; j < 16; ++j) f[j] = base[(kb + j) * stride + n];
  return f;
}

DEV v8f wmma_bf16(v16bf a, v16bf b, v8f c) {
  return __builtin_amdgcn_wmma_f32_16x16x32_bf16(false, a, false, b, (short)0, c,
                                                 false, false);
}

// ---------------------------------------------------------------------------
// Problem constants
// ---------------------------------------------------------------------------
static constexpr int NB   = 4;
static constexpr int LSEQ = 8192;
static constexpr int CC   = 256;
static constexpr int NL   = NB * LSEQ;      // 32768 rows
static constexpr int NHEAD = 8;

enum GemmEpi { EPI_ELU1 = 0, EPI_RELU = 1, EPI_LN_BF16 = 2, EPI_LN_RES = 3 };

// ---------------------------------------------------------------------------
// Elementwise conversion
// ---------------------------------------------------------------------------
__global__ void f32_to_bf16_kernel(const float* __restrict__ in,
                                   bf16_t* __restrict__ out, int n) {
  int i = blockIdx.x * 256 + threadIdx.x;
  if (i < n) out[i] = f2bf(in[i]);
}

// ---------------------------------------------------------------------------
// Generic bf16 WMMA GEMM:  Y[M,Nout] = epi( A[M,Ktot] @ W[Nout,Ktot]^T )
// Block: 256 threads (8 waves), BM=32, BN=256; wave grid 2 (M) x 4 (N).
// SPLIT: A column k<256 reads A0, k>=256 reads A1 (both row-major, ld=256).
// LN epilogues require Nout==256 and gridDim.y==1.
// ---------------------------------------------------------------------------
template <int EPI, bool SPLIT>
__global__ __launch_bounds__(256) void gemm_bf16_kernel(
    const bf16_t* __restrict__ A0, const bf16_t* __restrict__ A1,
    const bf16_t* __restrict__ W, int Ktot, int Nout,
    bf16_t* __restrict__ outB, float* __restrict__ outF,
    const float* __restrict__ gamma, const float* __restrict__ beta,
    const float* __restrict__ resid) {
  const int tid = threadIdx.x, lane = tid & 31, wave = tid >> 5;
  const int wm = wave & 1, wn = wave >> 1;
  const int rb = blockIdx.x * 32;
  const int cb = blockIdx.y * 256;
  const int rows0 = rb + wm * 16;
  const int cols0 = cb + wn * 64;

  v8f acc[4];
#pragma unroll
  for (int t = 0; t < 4; ++t) acc[t] = vzero8();

  for (int kk = 0; kk < Ktot; kk += 32) {
    const bf16_t* Asrc;
    int acol, lda;
    if (SPLIT) {
      lda = 256;
      if (kk < 256) { Asrc = A0; acol = kk; }
      else          { Asrc = A1; acol = kk - 256; }
    } else {
      Asrc = A0; acol = kk; lda = Ktot;
    }
    v16bf a = loadA_rm(Asrc, lda, rows0, acol);
#pragma unroll
    for (int t = 0; t < 4; ++t) {
      v16bf b = loadB_rm(W, Ktot, cols0 + t * 16, kk);
      acc[t] = wmma_bf16(a, b, acc[t]);
    }
  }

  const int hi = (lane >> 4) << 3;  // lanes 16..31 hold M rows +8

  if constexpr (EPI == EPI_ELU1 || EPI == EPI_RELU) {
#pragma unroll
    for (int t = 0; t < 4; ++t)
#pragma unroll
      for (int r = 0; r < 8; ++r) {
        int m = rb + wm * 16 + r + hi;
        int n = cb + wn * 64 + t * 16 + (lane & 15);
        float v = acc[t][r];
        if (EPI == EPI_ELU1) v = (v > 0.f) ? (v + 1.f) : __expf(v);  // elu(v)+1
        else                 v = fmaxf(v, 0.f);
        outB[(size_t)m * Nout + n] = f2bf(v);
      }
  } else {
    // Full-row LayerNorm epilogue (Nout == 256, one N block).
    __shared__ float sm[32 * 256];
    __shared__ float red[32][8][2];
#pragma unroll
    for (int t = 0; t < 4; ++t)
#pragma unroll
      for (int r = 0; r < 8; ++r) {
        int ml = wm * 16 + r + hi;
        int nl = wn * 64 + t * 16 + (lane & 15);
        sm[ml * 256 + nl] = acc[t][r];
      }
    __syncthreads();
    int row = tid >> 3, part = tid & 7;
    float s = 0.f, s2 = 0.f;
#pragma unroll
    for (int j = 0; j < 32; ++j) {
      float v = sm[row * 256 + part * 32 + j];
      s += v; s2 += v * v;
    }
    red[row][part][0] = s; red[row][part][1] = s2;
    __syncthreads();
    float S = 0.f, S2 = 0.f;
#pragma unroll
    for (int p = 0; p < 8; ++p) { S += red[row][p][0]; S2 += red[row][p][1]; }
    float mu  = S * (1.f / 256.f);
    float var = S2 * (1.f / 256.f) - mu * mu;
    float inv = rsqrtf(var + 1e-7f);
#pragma unroll
    for (int j = 0; j < 32; ++j) {
      int c = part * 32 + j;
      float v = (sm[row * 256 + c] - mu) * inv * gamma[c] + beta[c];
      size_t gi = (size_t)(rb + row) * 256 + c;
      if (EPI == EPI_LN_BF16) outB[gi] = f2bf(v);
      else                    outF[gi] = resid[gi] + v;
    }
  }
}

// ---------------------------------------------------------------------------
// K/V projection + per-head K^T·V partial accumulation (deterministic:
// per-block partials, reduced by kv_reduce_kernel).
// One block = 32 source rows. Phase 1: WMMA projections into LDS (K gets
// elu+1). Phase 2: per head h, KV_h += Ktile_h^T @ Vtile_h (32x32) via WMMA.
// ---------------------------------------------------------------------------
__global__ __launch_bounds__(256) void kv_accum_kernel(
    const bf16_t* __restrict__ srcb, const bf16_t* __restrict__ Wk,
    const bf16_t* __restrict__ Wv, float* __restrict__ KVpart,
    float* __restrict__ Kspart) {
  __shared__ bf16_t Kt[32 * 264];
  __shared__ bf16_t Vt[32 * 264];
  const int tid = threadIdx.x, lane = tid & 31, wave = tid >> 5;
  const int wm = wave & 1, wn = wave >> 1;
  const int rb = blockIdx.x * 32;
  const int rows0 = rb + wm * 16;
  const int cols0 = wn * 64;

  v8f aK[4], aV[4];
#pragma unroll
  for (int t = 0; t < 4; ++t) { aK[t] = vzero8(); aV[t] = vzero8(); }

  for (int kk = 0; kk < 256; kk += 32) {
    v16bf a = loadA_rm(srcb, 256, rows0, kk);
#pragma unroll
    for (int t = 0; t < 4; ++t) {
      v16bf bk = loadB_rm(Wk, 256, cols0 + t * 16, kk);
      aK[t] = wmma_bf16(a, bk, aK[t]);
      v16bf bv = loadB_rm(Wv, 256, cols0 + t * 16, kk);
      aV[t] = wmma_bf16(a, bv, aV[t]);
    }
  }

  const int hi = (lane >> 4) << 3;
#pragma unroll
  for (int t = 0; t < 4; ++t)
#pragma unroll
    for (int r = 0; r < 8; ++r) {
      int ml = wm * 16 + r + hi;
      int nl = cols0 + t * 16 + (lane & 15);
      float k = aK[t][r];
      k = (k > 0.f) ? (k + 1.f) : __expf(k);     // elu(k)+1
      Kt[ml * 264 + nl] = f2bf(k);
      Vt[ml * 264 + nl] = f2bf(aV[t][r]);
    }
  __syncthreads();

  // Ksum partial: column sums of elu(K)+1 over the 32 rows.
  {
    int c = tid;
    float s = 0.f;
#pragma unroll
    for (int r = 0; r < 32; ++r) s += bf2f(Kt[r * 264 + c]);
    Kspart[(size_t)blockIdx.x * 256 + c] = s;
  }

  // Phase 2: per-head KV (32x32) = Ktile_h^T (d x rows) @ Vtile_h (rows x d).
  const int head = wave;
#pragma unroll
  for (int sub = 0; sub < 4; ++sub) {
    int mt = sub >> 1, nt = sub & 1;
    v16bf a2 = loadA_km(Kt, 264, head * 32 + mt * 16);   // A[m,k'] = Kt[k'][..+m]
    v16bf b2 = loadB_km(Vt, 264, head * 32 + nt * 16);   // B[k',n] = Vt[k'][..+n]
    v8f acc = vzero8();
    acc = wmma_bf16(a2, b2, acc);
#pragma unroll
    for (int r = 0; r < 8; ++r) {
      int m = mt * 16 + r + hi;
      int n = nt * 16 + (lane & 15);
      KVpart[(size_t)blockIdx.x * 8192 + head * 1024 + m * 32 + n] = acc[r];
    }
  }
}

// Reduce per-block partials -> KV (bf16, [n,h,32,32]) and Ksum (f32, [n,256]).
__global__ __launch_bounds__(256) void kv_reduce_kernel(
    const float* __restrict__ KVpart, const float* __restrict__ Kspart,
    bf16_t* __restrict__ KVb, float* __restrict__ Ksum) {
  int i = blockIdx.x * 256 + threadIdx.x;
  if (i < NB * 8192) {
    int n = i >> 13, rest = i & 8191;
    const float* p = KVpart + (size_t)n * 256 * 8192 + rest;
    float s = 0.f;
    for (int b = 0; b < 256; ++b) s += p[(size_t)b * 8192];
    KVb[i] = f2bf(s);
  } else if (i < NB * 8192 + NB * 256) {
    int j = i - NB * 8192;
    int n = j >> 8, c = j & 255;
    const float* p = Kspart + (size_t)n * 256 * 256 + c;
    float s = 0.f;
    for (int b = 0; b < 256; ++b) s += p[b * 256];
    Ksum[j] = s;
  }
}

// ---------------------------------------------------------------------------
// Attention apply: msg0[l, h*32+e] = Z(l,h) * sum_d Q[l,h,d] * KV[h,d,e]
// One block = 32 query rows; wave w handles head w (2x2 WMMA tiles).
// ---------------------------------------------------------------------------
__global__ __launch_bounds__(256) void attn_kernel(
    const bf16_t* __restrict__ Qb, const bf16_t* __restrict__ KVb,
    const float* __restrict__ Ksum, bf16_t* __restrict__ msg0b) {
  __shared__ bf16_t KVs[NHEAD * 32 * 32];
  __shared__ float  Ks[256];
  __shared__ float  Zs[256];  // [row 0..31][head 0..7]
  const int tid = threadIdx.x, lane = tid & 31, wave = tid >> 5;
  const int rb = blockIdx.x * 32;
  const int nb = rb >> 13;

  for (int i = tid; i < NHEAD * 1024; i += 256) KVs[i] = KVb[(size_t)nb * 8192 + i];
  Ks[tid] = Ksum[nb * 256 + tid];
  __syncthreads();

  {
    int row = tid >> 3, head = tid & 7;
    float d = 0.f;
#pragma unroll
    for (int j = 0; j < 32; ++j)
      d += bf2f(Qb[(size_t)(rb + row) * 256 + head * 32 + j]) * Ks[head * 32 + j];
    Zs[row * 8 + head] = 1.f / (d + 1e-6f);
  }
  __syncthreads();

  const int head = wave;
  const int hi = (lane >> 4) << 3;
#pragma unroll
  for (int sub = 0; sub < 4; ++sub) {
    int mt = sub >> 1, nt = sub & 1;
    v16bf a = loadA_rm(Qb, 256, rb + mt * 16, head * 32);
    v16bf b = loadB_km(KVs + head * 1024, 32, nt * 16);
    v8f acc = vzero8();
    acc = wmma_bf16(a, b, acc);
#pragma unroll
    for (int r = 0; r < 8; ++r) {
      int ml = mt * 16 + r + hi;
      float v = acc[r] * Zs[ml * 8 + head];
      msg0b[(size_t)(rb + ml) * 256 + head * 32 + nt * 16 + (lane & 15)] = f2bf(v);
    }
  }
}

// ---------------------------------------------------------------------------
// Host launcher
// ---------------------------------------------------------------------------
extern "C" void kernel_launch(void* const* d_in, const int* in_sizes, int n_in,
                              void* d_out, int out_size, void* d_ws, size_t ws_size,
                              hipStream_t stream) {
  (void)in_sizes; (void)n_in; (void)out_size; (void)ws_size;

  const float* x    = (const float*)d_in[0];
  const float* src  = (const float*)d_in[1];
  const float* Wq   = (const float*)d_in[2];
  const float* Wk   = (const float*)d_in[3];
  const float* Wv   = (const float*)d_in[4];
  const float* Wmg  = (const float*)d_in[5];
  const float* Wm1  = (const float*)d_in[6];
  const float* Wm2  = (const float*)d_in[7];
  const float* ln1g = (const float*)d_in[8];
  const float* ln1b = (const float*)d_in[9];
  const float* ln2g = (const float*)d_in[10];
  const float* ln2b = (const float*)d_in[11];
  float* out = (float*)d_out;

  char* ws = (char*)d_ws;
  size_t off = 0;
  auto alloc = [&](size_t bytes) -> char* {
    char* p = ws + off;
    off = (off + bytes + 255) & ~(size_t)255;
    return p;
  };

  bf16_t* xb    = (bf16_t*)alloc((size_t)NL * CC * 2);
  bf16_t* srcb  = (bf16_t*)alloc((size_t)NL * CC * 2);
  bf16_t* Qb    = (bf16_t*)alloc((size_t)NL * CC * 2);
  bf16_t* msgb  = (bf16_t*)alloc((size_t)NL * CC * 2);
  bf16_t* Wqb   = (bf16_t*)alloc((size_t)CC * CC * 2);
  bf16_t* Wkb   = (bf16_t*)alloc((size_t)CC * CC * 2);
  bf16_t* Wvb   = (bf16_t*)alloc((size_t)CC * CC * 2);
  bf16_t* Wmgb  = (bf16_t*)alloc((size_t)CC * CC * 2);
  bf16_t* Wm1b  = (bf16_t*)alloc((size_t)512 * 512 * 2);
  bf16_t* Wm2b  = (bf16_t*)alloc((size_t)CC * 512 * 2);
  bf16_t* KVb   = (bf16_t*)alloc((size_t)NB * NHEAD * 32 * 32 * 2);
  float*  KsumF = (float*) alloc((size_t)NB * CC * 4);
  float*  Kspart= (float*) alloc((size_t)1024 * 256 * 4);
  // Aliased region: KVpart (used by kernels 3-4) then reused as h (kernels 7-8).
  char*   big   = alloc((size_t)1024 * 8192 * 4);       // 32 MB
  float*  KVpart= (float*)big;
  bf16_t* hb    = (bf16_t*)big;                         // [NL, 512] bf16 = 32 MB
  // msg0 aliases srcb's slot? srcb still needed by kv_accum before attn runs,
  // and attn (producer of msg0) runs after kv_accum -> safe to reuse srcb.
  bf16_t* msg0b = srcb;

  auto cvt = [&](const float* in, bf16_t* o, size_t n) {
    f32_to_bf16_kernel<<<dim3((unsigned)((n + 255) / 256)), 256, 0, stream>>>(in, o, (int)n);
  };

  // 1) conversions
  cvt(x,   xb,   (size_t)NL * CC);
  cvt(src, srcb, (size_t)NL * CC);
  cvt(Wq,  Wqb,  (size_t)CC * CC);
  cvt(Wk,  Wkb,  (size_t)CC * CC);
  cvt(Wv,  Wvb,  (size_t)CC * CC);
  cvt(Wmg, Wmgb, (size_t)CC * CC);
  cvt(Wm1, Wm1b, (size_t)512 * 512);
  cvt(Wm2, Wm2b, (size_t)CC * 512);

  const int MB = NL / 32;  // 1024 row-blocks

  // 2) Q = elu(x @ Wq^T) + 1
  gemm_bf16_kernel<EPI_ELU1, false><<<dim3(MB, 1), 256, 0, stream>>>(
      xb, nullptr, Wqb, 256, 256, Qb, nullptr, nullptr, nullptr, nullptr);

  // 3) K/V projection + per-block KV / Ksum partials
  kv_accum_kernel<<<dim3(MB), 256, 0, stream>>>(srcb, Wkb, Wvb, KVpart, Kspart);

  // 4) reduce partials
  kv_reduce_kernel<<<dim3((NB * 8192 + NB * 256 + 255) / 256), 256, 0, stream>>>(
      KVpart, Kspart, KVb, KsumF);

  // 5) attention apply -> msg0 (bf16, overwrites srcb slot)
  attn_kernel<<<dim3(MB), 256, 0, stream>>>(Qb, KVb, KsumF, msg0b);

  // 6) msg = LN1(msg0 @ Wmerge^T)   (bf16)
  gemm_bf16_kernel<EPI_LN_BF16, false><<<dim3(MB, 1), 256, 0, stream>>>(
      msg0b, nullptr, Wmgb, 256, 256, msgb, nullptr, ln1g, ln1b, nullptr);

  // 7) h = relu(concat(x, msg) @ Wmlp1^T)  (bf16, [NL,512]) -- reuses KVpart mem
  gemm_bf16_kernel<EPI_RELU, true><<<dim3(MB, 2), 256, 0, stream>>>(
      xb, msgb, Wm1b, 512, 512, hb, nullptr, nullptr, nullptr, nullptr);

  // 8) out = x + LN2(h @ Wmlp2^T)   (f32)
  gemm_bf16_kernel<EPI_LN_RES, false><<<dim3(MB, 1), 256, 0, stream>>>(
      hb, nullptr, Wm2b, 512, 256, nullptr, out, ln2g, ln2b, x);
}
